// ActorHead_61607010893924
// MI455X (gfx1250) — matrix operand order repaired and checked by reading, not compiled, not run on hardware
//
#include <hip/hip_runtime.h>
#include <hip/hip_bf16.h>
#include <stdint.h>

// ---------------------------------------------------------------------------
// MI455X (gfx1250) wave32 bf16-WMMA attention pipeline.
// ---------------------------------------------------------------------------

typedef __attribute__((ext_vector_type(16))) __bf16 v16bf;
typedef __attribute__((ext_vector_type(8)))  float  v8f;
typedef __attribute__((ext_vector_type(4)))  int    v4i;

#define D_MODEL 512
#define NHEAD   4
#define DKH     128
#define NACT    4096
#define NBV     4096
#define NEGF    (-1.0e9f)

#if defined(__has_builtin)
#if __has_builtin(__builtin_amdgcn_global_load_async_to_lds_b128)
#define HAVE_ASYNC_B128 1
#endif
#if __has_builtin(__builtin_amdgcn_s_wait_asynccnt)
#define HAVE_WAIT_ASYNC 1
#endif
#endif

// nan_to_num(nan->0, +inf->1, -inf->-1), branchless (pure v_cndmask selects).
__device__ __forceinline__ float ntn(float x) {
  unsigned int u = __float_as_uint(x);
  unsigned int a = u & 0x7FFFFFFFu;
  float infv = (u & 0x80000000u) ? -1.0f : 1.0f;
  float r = (a == 0x7F800000u) ? infv : x;
  return (a > 0x7F800000u) ? 0.0f : r;
}

// float -> bf16 (round to nearest even); inputs are pre-sanitized
__device__ __forceinline__ unsigned short f2bf(float f) {
  unsigned int u = __float_as_uint(f);
  unsigned int r = u + 0x7FFFu + ((u >> 16) & 1u);
  return (unsigned short)(r >> 16);
}

struct __align__(16) U128 { unsigned int x, y, z, w; };
struct BF16x16 { U128 lo, hi; };

// Build a v16bf WMMA fragment from two 16-byte LDS segments.
__device__ __forceinline__ v16bf ld_frag(const unsigned short* p0,
                                         const unsigned short* p1) {
  BF16x16 t;
  t.lo = *reinterpret_cast<const U128*>(p0);
  t.hi = *reinterpret_cast<const U128*>(p1);
  return __builtin_bit_cast(v16bf, t);
}

__device__ __forceinline__ v8f wmma_bf16(v16bf a, v16bf b, v8f c) {
  return __builtin_amdgcn_wmma_f32_16x16x32_bf16(false, a, false, b,
                                                 (short)0, c, false, false);
}

// 16-byte global -> LDS copy: async (ASYNCcnt-tracked, no VGPR bounce) when
// the toolchain exposes the gfx1250 builtin, else a plain VGPR copy.
__device__ __forceinline__ void copy16_g2l(unsigned short* ldst,
                                           const unsigned short* gsrc) {
#ifdef HAVE_ASYNC_B128
  __builtin_amdgcn_global_load_async_to_lds_b128(
      (__attribute__((address_space(1))) v4i*)gsrc,
      (__attribute__((address_space(3))) v4i*)ldst, 0, 0);
#else
  *reinterpret_cast<U128*>(ldst) = *reinterpret_cast<const U128*>(gsrc);
#endif
}

__device__ __forceinline__ void async_fence() {
#ifdef HAVE_ASYNC_B128
#ifdef HAVE_WAIT_ASYNC
  __builtin_amdgcn_s_wait_asynccnt(0);
#else
  asm volatile("s_wait_asynccnt 0" ::: "memory");
#endif
#endif
}

// ---------------------------------------------------------------------------
// K0: fold out_proj into a 128x512 "topic" weight: W2 = mean_h out_w[h*128+j]
// ---------------------------------------------------------------------------
__global__ __launch_bounds__(256) void prep_kernel(
    const float* __restrict__ out_w, const float* __restrict__ out_b,
    unsigned short* __restrict__ W2, float* __restrict__ b2) {
  int idx = blockIdx.x * 256 + threadIdx.x;           // 128*512 total
  int j = idx >> 9, d = idx & 511;
  float s = 0.25f * (out_w[(size_t)j * 512 + d] +
                     out_w[(size_t)(128 + j) * 512 + d] +
                     out_w[(size_t)(256 + j) * 512 + d] +
                     out_w[(size_t)(384 + j) * 512 + d]);
  W2[(size_t)j * 512 + d] = f2bf(s);
  if (d == 0)
    b2[j] = 0.25f * (out_b[j] + out_b[128 + j] + out_b[256 + j] + out_b[384 + j]);
}

__global__ void zero_kernel(float* __restrict__ p, int n) {
  int i = blockIdx.x * blockDim.x + threadIdx.x;
  if (i < n) p[i] = 0.0f;
}

// ---------------------------------------------------------------------------
// K1: QKV projection.  C[4096,512] = ntn(A) @ W^T + b, stored bf16.
// grid = (rowblk 32, colblk 4, which 3), block = 256 (8 waves, 16 rows each)
// ---------------------------------------------------------------------------
__global__ __launch_bounds__(256) void proj_kernel(
    const float* __restrict__ a_z, const float* __restrict__ bv_z,
    const float* __restrict__ ipw, const float* __restrict__ ipb,
    unsigned short* __restrict__ Qb, unsigned short* __restrict__ Kb,
    unsigned short* __restrict__ Vb) {
  __shared__ __align__(16) unsigned short aT[128 * 32];
  __shared__ __align__(16) unsigned short wT[128 * 32];
  const int tid = threadIdx.x;
  const int wave = tid >> 5, lane = tid & 31, half = lane >> 4, ln = lane & 15;
  const int rb = blockIdx.x, cb = blockIdx.y, z = blockIdx.z;
  const float* A = (z == 0) ? a_z : bv_z;
  const float* W = ipw + (size_t)z * D_MODEL * D_MODEL;
  const float* bias = ipb + z * D_MODEL;
  unsigned short* Out = (z == 0) ? Qb : (z == 1) ? Kb : Vb;
  const float scale = (z == 0) ? 0.08838834764831845f : 1.0f;  // 1/sqrt(128)

  v8f acc[8];
#pragma unroll
  for (int nt = 0; nt < 8; ++nt)
#pragma unroll
    for (int g = 0; g < 8; ++g) acc[nt][g] = 0.0f;

  for (int k0 = 0; k0 < D_MODEL; k0 += 32) {
#pragma unroll
    for (int i = 0; i < 4; ++i) {
      int cc = tid + i * 256;
      int row = cc >> 3, kc = (cc & 7) * 4;
      float4 av = *reinterpret_cast<const float4*>(
          &A[(size_t)(rb * 128 + row) * D_MODEL + k0 + kc]);
      uint2 pa;
      pa.x = (unsigned)f2bf(ntn(av.x)) | ((unsigned)f2bf(ntn(av.y)) << 16);
      pa.y = (unsigned)f2bf(ntn(av.z)) | ((unsigned)f2bf(ntn(av.w)) << 16);
      *reinterpret_cast<uint2*>(&aT[row * 32 + kc]) = pa;
      float4 wv = *reinterpret_cast<const float4*>(
          &W[(size_t)(cb * 128 + row) * D_MODEL + k0 + kc]);
      uint2 pw;
      pw.x = (unsigned)f2bf(wv.x) | ((unsigned)f2bf(wv.y) << 16);
      pw.y = (unsigned)f2bf(wv.z) | ((unsigned)f2bf(wv.w) << 16);
      *reinterpret_cast<uint2*>(&wT[row * 32 + kc]) = pw;
    }
    __syncthreads();
    // prefetch next k-slice (one 128B line per thread per matrix)
    if (k0 + 32 < D_MODEL) {
      int prow = tid >> 1;
      int pcol = k0 + 32;
      __builtin_prefetch(&A[(size_t)(rb * 128 + prow) * D_MODEL + pcol], 0, 1);
      __builtin_prefetch(&W[(size_t)(cb * 128 + prow) * D_MODEL + pcol], 0, 1);
    }
    const unsigned short* ap = &aT[(wave * 16 + ln) * 32 + half * 8];
    v16bf af = ld_frag(ap, ap + 16);
    // double-buffered B fragments: issue nt+1 loads before WMMA nt
    const unsigned short* bp0 = &wT[(ln) * 32 + half * 8];
    v16bf bcur = ld_frag(bp0, bp0 + 16);
#pragma unroll
    for (int nt = 0; nt < 8; ++nt) {
      v16bf bnxt = bcur;
      if (nt < 7) {
        const unsigned short* bp = &wT[((nt + 1) * 16 + ln) * 32 + half * 8];
        bnxt = ld_frag(bp, bp + 16);
      }
      acc[nt] = wmma_bf16(af, bcur, acc[nt]);
      bcur = bnxt;
    }
    __syncthreads();
  }
#pragma unroll
  for (int nt = 0; nt < 8; ++nt) {
    int col = cb * 128 + nt * 16 + ln;
    float bv = bias[col];
#pragma unroll
    for (int g = 0; g < 8; ++g) {
      int row = rb * 128 + wave * 16 + g + 8 * half;
      Out[(size_t)row * D_MODEL + col] = f2bf((acc[nt][g] + bv) * scale);
    }
  }
}

// ---------------------------------------------------------------------------
// K2: per-row stats of bias = where(mask, ntn(weight), -1e9):
//     rowmax, rowsum(exp(b-max)), valid.  One wave per row, branchless.
// ---------------------------------------------------------------------------
__global__ __launch_bounds__(256) void bias_stats_kernel(
    const unsigned char* __restrict__ mask, const float* __restrict__ weight,
    float* __restrict__ rowmax, float* __restrict__ rowsum,
    float* __restrict__ valid) {
  int wave = threadIdx.x >> 5, lane = threadIdx.x & 31;
  int row = blockIdx.x * 8 + wave;
  const float* wrow = weight + (size_t)row * NBV;
  const unsigned char* mrow = mask + (size_t)row * NBV;
  float mx = -3.0e38f, sm = 0.0f, anyf = 0.0f;
  for (int c = lane; c < NBV; c += 32) {
    unsigned char m = mrow[c];
    float b = m ? ntn(wrow[c]) : NEGF;
    anyf = fmaxf(anyf, m ? 1.0f : 0.0f);
    float nm = fmaxf(mx, b);
    sm = sm * __expf(mx - nm) + __expf(b - nm);
    mx = nm;
  }
#pragma unroll
  for (int off = 16; off > 0; off >>= 1) {
    float om = __shfl_xor(mx, off);
    float os = __shfl_xor(sm, off);
    float oa = __shfl_xor(anyf, off);
    float nm = fmaxf(mx, om);
    sm = sm * __expf(mx - nm) + os * __expf(om - nm);
    mx = nm;
    anyf = fmaxf(anyf, oa);
  }
  if (lane == 0) { rowmax[row] = mx; rowsum[row] = sm; valid[row] = anyf; }
}

// ---------------------------------------------------------------------------
// K3: flash attention per (64-query block, head).  block = 128 (4 waves,
// 16 query rows each).  Online softmax; influence accumulated online.
// grid = (4096/64, NHEAD)
// ---------------------------------------------------------------------------
__global__ __launch_bounds__(128) void attn_kernel(
    const unsigned short* __restrict__ Qb, const unsigned short* __restrict__ Kb,
    const unsigned short* __restrict__ Vb,
    const unsigned char* __restrict__ mask, const float* __restrict__ weight,
    const float* __restrict__ rowmax, const float* __restrict__ rowsum,
    unsigned short* __restrict__ ctx, float* __restrict__ infl) {
  __shared__ __align__(16) unsigned short sQ[64 * 128];
  __shared__ __align__(16) unsigned short sK[64 * 128];
  __shared__ __align__(16) unsigned short sVt[128 * 72];   // padded stride 72
  __shared__ __align__(16) float          sB[64 * 64];
  __shared__ __align__(16) unsigned short sP[4 * 16 * 64]; // per-wave P tiles

  const int tid = threadIdx.x;
  const int wave = tid >> 5, lane = tid & 31, half = lane >> 4, ln = lane & 15;
  const int q0 = blockIdx.x * 64;
  const int h = blockIdx.y;

  // Q tile [64 q][128 dk] (bf16), 16B chunks (async path when available)
#pragma unroll
  for (int i = 0; i < 8; ++i) {
    int cc = tid + i * 128;
    int row = cc >> 4, kc = (cc & 15) * 8;
    copy16_g2l(&sQ[row * 128 + kc],
               &Qb[(size_t)(q0 + row) * D_MODEL + h * DKH + kc]);
  }
  async_fence();
  __syncthreads();

  v16bf qf[4];
#pragma unroll
  for (int kk = 0; kk < 4; ++kk) {
    const unsigned short* p0 = &sQ[(wave * 16 + ln) * 128 + kk * 32 + half * 8];
    qf[kk] = ld_frag(p0, p0 + 16);
  }

  float ms[8], ls[8], ib[8], mb[8], zb[8];
  v8f O[8];
#pragma unroll
  for (int g = 0; g < 8; ++g) {
    int row = q0 + wave * 16 + g + 8 * half;
    ms[g] = -3.0e38f; ls[g] = 0.0f; ib[g] = 0.0f;
    mb[g] = rowmax[row]; zb[g] = rowsum[row];
#pragma unroll
    for (int dt = 0; dt < 8; ++dt) O[dt][g] = 0.0f;
  }

  unsigned short* pw = &sP[wave * 16 * 64];

  for (int k0 = 0; k0 < NBV; k0 += 64) {
    // K tile (row-major [key][dk]) in 16B chunks (async path when available)
#pragma unroll
    for (int i = 0; i < 8; ++i) {
      int cc = tid + i * 128;
      int row = cc >> 4, kc = (cc & 15) * 8;
      copy16_g2l(&sK[row * 128 + kc],
                 &Kb[(size_t)(k0 + row) * D_MODEL + h * DKH + kc]);
    }
    // V tile transposed ([dk][key], stride 72)
#pragma unroll
    for (int i = 0; i < 16; ++i) {
      int cc = tid + i * 128;
      int row = cc >> 5, kc = (cc & 31) * 4;
      uint2 vv = *reinterpret_cast<const uint2*>(
          &Vb[(size_t)(k0 + row) * D_MODEL + h * DKH + kc]);
      sVt[(kc + 0) * 72 + row] = (unsigned short)(vv.x & 0xFFFFu);
      sVt[(kc + 1) * 72 + row] = (unsigned short)(vv.x >> 16);
      sVt[(kc + 2) * 72 + row] = (unsigned short)(vv.y & 0xFFFFu);
      sVt[(kc + 3) * 72 + row] = (unsigned short)(vv.y >> 16);
    }
    // bias tile [64 q][64 key], mask folded in (masked = -1e9)
#pragma unroll
    for (int i = 0; i < 8; ++i) {
      int cc = tid + i * 128;
      int row = cc >> 4, c4 = (cc & 15) * 4;
      float4 wv = *reinterpret_cast<const float4*>(
          &weight[(size_t)(q0 + row) * NBV + k0 + c4]);
      const unsigned char* mp = &mask[(size_t)(q0 + row) * NBV + k0 + c4];
      float4 bv;
      bv.x = mp[0] ? ntn(wv.x) : NEGF;
      bv.y = mp[1] ? ntn(wv.y) : NEGF;
      bv.z = mp[2] ? ntn(wv.z) : NEGF;
      bv.w = mp[3] ? ntn(wv.w) : NEGF;
      *reinterpret_cast<float4*>(&sB[row * 64 + c4]) = bv;
    }
    async_fence();
    __syncthreads();

    // prefetch next key block while this one computes (global_prefetch_b8)
    if (k0 + 64 < NBV) {
      int prow = tid >> 1;                       // 64 rows x 2 cache lines
      int pseg = (tid & 1) * 64;
      const unsigned short* nk =
          &Kb[(size_t)(k0 + 64 + prow) * D_MODEL + h * DKH + pseg];
      const unsigned short* nv =
          &Vb[(size_t)(k0 + 64 + prow) * D_MODEL + h * DKH + pseg];
      const float* nw = &weight[(size_t)(q0 + prow) * NBV + k0 + 64 + pseg / 2];
      __builtin_prefetch(nk, 0, 1);
      __builtin_prefetch(nv, 0, 1);
      __builtin_prefetch(nw, 0, 1);
    }

    // S = Q K^T + bias  (wave: 16 rows x 64 keys = 4 C tiles),
    // double-buffered K fragments.
    float sc[4][8], bb[4][8];
#pragma unroll
    for (int nt = 0; nt < 4; ++nt) {
      const unsigned short* bp0 = &sK[(nt * 16 + ln) * 128 + half * 8];
      v16bf bcur = ld_frag(bp0, bp0 + 16);
      v8f acc = {0, 0, 0, 0, 0, 0, 0, 0};
#pragma unroll
      for (int kk = 0; kk < 4; ++kk) {
        v16bf bnxt = bcur;
        if (kk < 3) {
          const unsigned short* bp =
              &sK[(nt * 16 + ln) * 128 + (kk + 1) * 32 + half * 8];
          bnxt = ld_frag(bp, bp + 16);
        }
        acc = wmma_bf16(qf[kk], bcur, acc);
        bcur = bnxt;
      }
#pragma unroll
      for (int g = 0; g < 8; ++g) {
        float b = sB[(wave * 16 + g + 8 * half) * 64 + nt * 16 + ln];
        bb[nt][g] = b;
        sc[nt][g] = acc[g] + b;
      }
    }

    // online softmax per row (16-lane xor reductions within each half)
#pragma unroll
    for (int g = 0; g < 8; ++g) {
      float rm = fmaxf(fmaxf(sc[0][g], sc[1][g]), fmaxf(sc[2][g], sc[3][g]));
#pragma unroll
      for (int off = 8; off > 0; off >>= 1) rm = fmaxf(rm, __shfl_xor(rm, off));
      float mnew = fmaxf(ms[g], rm);
      float cs = __expf(ms[g] - mnew);
      ms[g] = mnew;
      ls[g] *= cs; ib[g] *= cs;
#pragma unroll
      for (int dt = 0; dt < 8; ++dt) O[dt][g] *= cs;
      float psum = 0.0f, epsum = 0.0f;
#pragma unroll
      for (int nt = 0; nt < 4; ++nt) {
        float p = __expf(sc[nt][g] - mnew);
        sc[nt][g] = p;
        float eb = (bb[nt][g] > -1.0e8f) ? __expf(bb[nt][g] - mb[g]) : 0.0f;
        psum += p; epsum += eb * p;
      }
#pragma unroll
      for (int off = 8; off > 0; off >>= 1) {
        psum += __shfl_xor(psum, off);
        epsum += __shfl_xor(epsum, off);
      }
      ls[g] += psum; ib[g] += epsum;
    }

    // stage P as bf16 in per-wave LDS region (C layout -> A layout)
#pragma unroll
    for (int nt = 0; nt < 4; ++nt)
#pragma unroll
      for (int g = 0; g < 8; ++g)
        pw[(g + 8 * half) * 64 + nt * 16 + ln] = f2bf(sc[nt][g]);

    // P fragments depend only on the K-chunk: load once, reuse for all dt
    v16bf pa[2];
#pragma unroll
    for (int kk = 0; kk < 2; ++kk) {
      const unsigned short* ap = &pw[ln * 64 + kk * 32 + half * 8];
      pa[kk] = ld_frag(ap, ap + 16);
    }

    // O += P @ V, double-buffered V fragments
    {
      const unsigned short* bp0 = &sVt[(ln)*72 + half * 8];
      v16bf bcur = ld_frag(bp0, bp0 + 16);
#pragma unroll
      for (int dt = 0; dt < 8; ++dt) {
        v8f acc = O[dt];
#pragma unroll
        for (int kk = 0; kk < 2; ++kk) {
          v16bf bnxt = bcur;
          if (!(dt == 7 && kk == 1)) {
            int ndt = (kk == 0) ? dt : dt + 1;
            int nkk = (kk == 0) ? 1 : 0;
            const unsigned short* bp =
                &sVt[(ndt * 16 + ln) * 72 + nkk * 32 + half * 8];
            bnxt = ld_frag(bp, bp + 16);
          }
          acc = wmma_bf16(pa[kk], bcur, acc);
          bcur = bnxt;
        }
        O[dt] = acc;
      }
    }
    __syncthreads();
  }

  // epilogue: normalize, write ctx (bf16), accumulate influence
#pragma unroll
  for (int g = 0; g < 8; ++g) {
    int row = q0 + wave * 16 + g + 8 * half;
    float inv = 1.0f / ls[g];
#pragma unroll
    for (int dt = 0; dt < 8; ++dt)
      ctx[(size_t)row * D_MODEL + h * DKH + dt * 16 + ln] = f2bf(O[dt][g] * inv);
    if (ln == g)
      atomicAdd(&infl[row], ib[g] / (ls[g] * zb[g]) * 0.25f);
  }
}

// ---------------------------------------------------------------------------
// K4: topic = valid ? ctx @ W2^T + b2 : 0    (4096 x 128 x 512, bf16 WMMA)
// grid = 32, block = 256
// ---------------------------------------------------------------------------
__global__ __launch_bounds__(256) void topic_kernel(
    const unsigned short* __restrict__ ctx, const unsigned short* __restrict__ W2,
    const float* __restrict__ b2, const float* __restrict__ valid,
    float* __restrict__ topic) {
  __shared__ __align__(16) unsigned short aT[128 * 32];
  __shared__ __align__(16) unsigned short wT[128 * 32];
  const int tid = threadIdx.x;
  const int wave = tid >> 5, lane = tid & 31, half = lane >> 4, ln = lane & 15;
  const int rb = blockIdx.x;

  v8f acc[8];
#pragma unroll
  for (int nt = 0; nt < 8; ++nt)
#pragma unroll
    for (int g = 0; g < 8; ++g) acc[nt][g] = 0.0f;

  for (int k0 = 0; k0 < D_MODEL; k0 += 32) {
    // pure bf16 copies: async path when available (16B chunks)
#pragma unroll
    for (int i = 0; i < 2; ++i) {
      int cc = tid + i * 256;
      int row = cc >> 2, kc = (cc & 3) * 8;
      copy16_g2l(&aT[row * 32 + kc],
                 &ctx[(size_t)(rb * 128 + row) * D_MODEL + k0 + kc]);
      copy16_g2l(&wT[row * 32 + kc],
                 &W2[(size_t)row * D_MODEL + k0 + kc]);
    }
    async_fence();
    __syncthreads();
    const unsigned short* ap = &aT[(wave * 16 + ln) * 32 + half * 8];
    v16bf af = ld_frag(ap, ap + 16);
    const unsigned short* bp0 = &wT[(ln) * 32 + half * 8];
    v16bf bcur = ld_frag(bp0, bp0 + 16);
#pragma unroll
    for (int nt = 0; nt < 8; ++nt) {
      v16bf bnxt = bcur;
      if (nt < 7) {
        const unsigned short* bp = &wT[((nt + 1) * 16 + ln) * 32 + half * 8];
        bnxt = ld_frag(bp, bp + 16);
      }
      acc[nt] = wmma_bf16(af, bcur, acc[nt]);
      bcur = bnxt;
    }
    __syncthreads();
  }
#pragma unroll
  for (int nt = 0; nt < 8; ++nt) {
    int col = nt * 16 + ln;
    float bv = b2[col];
#pragma unroll
    for (int g = 0; g < 8; ++g) {
      int row = rb * 128 + wave * 16 + g + 8 * half;
      topic[(size_t)row * DKH + col] =
          (valid[row] > 0.5f) ? (acc[nt][g] + bv) : 0.0f;
    }
  }
}

// ---------------------------------------------------------------------------
extern "C" void kernel_launch(void* const* d_in, const int* in_sizes, int n_in,
                              void* d_out, int out_size, void* d_ws,
                              size_t ws_size, hipStream_t stream) {
  const float* a_z  = (const float*)d_in[0];
  const float* bv_z = (const float*)d_in[1];
  const unsigned char* mask = (const unsigned char*)d_in[2];
  const float* weight = (const float*)d_in[3];
  const float* ipw = (const float*)d_in[4];
  const float* ipb = (const float*)d_in[5];
  const float* out_w = (const float*)d_in[6];
  const float* out_b = (const float*)d_in[7];

  float* topic = (float*)d_out;                       // [4096,128]
  float* infl  = topic + (size_t)NACT * DKH;          // [4096]

  char* ws = (char*)d_ws;
  unsigned short* Qb = (unsigned short*)ws; ws += (size_t)NACT * D_MODEL * 2;
  unsigned short* Kb = (unsigned short*)ws; ws += (size_t)NBV * D_MODEL * 2;
  unsigned short* Vb = (unsigned short*)ws; ws += (size_t)NBV * D_MODEL * 2;
  unsigned short* ctx = (unsigned short*)ws; ws += (size_t)NACT * D_MODEL * 2;
  unsigned short* W2 = (unsigned short*)ws; ws += (size_t)DKH * D_MODEL * 2;
  float* b2 = (float*)ws;     ws += DKH * 4;
  float* rowmax = (float*)ws; ws += NACT * 4;
  float* rowsum = (float*)ws; ws += NACT * 4;
  float* valid = (float*)ws;  ws += NACT * 4;

  prep_kernel<<<(DKH * D_MODEL) / 256, 256, 0, stream>>>(out_w, out_b, W2, b2);
  zero_kernel<<<NACT / 256, 256, 0, stream>>>(infl, NACT);
  proj_kernel<<<dim3(NACT / 128, D_MODEL / 128, 3), 256, 0, stream>>>(
      a_z, bv_z, ipw, ipb, Qb, Kb, Vb);
  bias_stats_kernel<<<NACT / 8, 256, 0, stream>>>(mask, weight, rowmax, rowsum,
                                                  valid);
  attn_kernel<<<dim3(NACT / 64, NHEAD), 128, 0, stream>>>(
      Qb, Kb, Vb, mask, weight, rowmax, rowsum, ctx, infl);
  topic_kernel<<<NACT / 128, 256, 0, stream>>>(ctx, W2, b2, valid, topic);
}
